// RawDotProductAttention_79413945303402
// MI455X (gfx1250) — matrix operand order repaired and checked by reading, not compiled, not run on hardware
//
#include <hip/hip_runtime.h>

// Flat flash-attention: M = B*S = 8192 query rows, N = 8192 keys, D = 128.
#define BQ 16            // query rows per wave (WMMA M)
#define BC 64            // keys per main-loop iteration
#define WAVES 4
#define BLOCK (WAVES * 32)
#define BR (WAVES * BQ)  // 64 query rows per workgroup
#define DH 128
#define MTOT 8192
#define KPAD 136         // bf16 elems per sK row (128 + 8 pad), 272B (16B aligned)
#define VPAD 72          // bf16 elems per sVT row (64 + 8 pad), 144B (16B aligned)

#if __has_builtin(__builtin_amdgcn_tensor_load_to_lds)
#define HAS_TDM 1
#else
#define HAS_TDM 0
#endif

typedef __attribute__((ext_vector_type(16))) __bf16 v16bf;
typedef __attribute__((ext_vector_type(8)))  float  v8f;
typedef __attribute__((ext_vector_type(4)))  unsigned int u32x4;
typedef __attribute__((ext_vector_type(8)))  int    i32x8;
typedef __attribute__((ext_vector_type(4)))  int    i32x4;
typedef __attribute__((ext_vector_type(4)))  float  f32x4;
typedef __attribute__((ext_vector_type(2)))  float  f32x2;
typedef __attribute__((ext_vector_type(2)))  __bf16 bf16x2;

union Frag16 { unsigned short h[16]; u32x4 q[2]; v16bf v; };

// packed f32 pair -> packed bf16 pair (selects v_cvt_pk_bf16_f32)
__device__ __forceinline__ unsigned int pack2(float a, float b) {
  f32x2 f = {a, b};
  bf16x2 h = __builtin_convertvector(f, bf16x2);
  return __builtin_bit_cast(unsigned int, h);
}
__device__ __forceinline__ unsigned short bfu(float a) {
  return __builtin_bit_cast(unsigned short, (__bf16)a);
}
// VALU xor-shuffle within each 16-lane row via v_permlane16_b32
__device__ __forceinline__ float lanexor(float x, unsigned lo, unsigned hi) {
  unsigned u = __builtin_bit_cast(unsigned, x);
  unsigned r = __builtin_amdgcn_permlane16(u, u, lo, hi, false, false);
  return __builtin_bit_cast(float, r);
}
#define XOR1_LO 0x67452301u
#define XOR1_HI 0xEFCDAB89u
#define XOR2_LO 0x54761032u
#define XOR2_HI 0xDCFE98BAu
#define XOR4_LO 0x32107654u
#define XOR4_HI 0xBA98FEDCu
#define XOR8_LO 0xFEDCBA98u
#define XOR8_HI 0x76543210u

__device__ __forceinline__ float rowmax16(float t) {
  t = fmaxf(t, lanexor(t, XOR1_LO, XOR1_HI));
  t = fmaxf(t, lanexor(t, XOR2_LO, XOR2_HI));
  t = fmaxf(t, lanexor(t, XOR4_LO, XOR4_HI));
  t = fmaxf(t, lanexor(t, XOR8_LO, XOR8_HI));
  return t;
}
__device__ __forceinline__ float rowsum16(float t) {
  t += lanexor(t, XOR1_LO, XOR1_HI);
  t += lanexor(t, XOR2_LO, XOR2_HI);
  t += lanexor(t, XOR4_LO, XOR4_HI);
  t += lanexor(t, XOR8_LO, XOR8_HI);
  return t;
}

#if HAS_TDM
// TDM DMA: 64x128 fp32 tile (rows stride 128) from global -> LDS (linear).
// D# per CDNA5 ISA ch.8: group0 = {count=1, lds_addr, global_addr[56:0], type=2},
// group1 = {data_size=4B, tensor_dim0=128, tensor_dim1=8192, tile=128x64,
//           dim0_stride=128}. 2D tile -> remaining descriptor groups zero.
// This toolchain exposes the 6-arg builtin:
//   (uint32x4 g0, int32x8 g1, int32x4, int32x4, int32x8, i32 cpol)
__device__ __forceinline__ void tdm_load_tile_f32(const float* g, unsigned lds_off) {
  unsigned long long ga = (unsigned long long)(const void*)g;
  u32x4 g0;
  g0[0] = 1u;                                              // count=1, no gather
  g0[1] = lds_off;                                         // lds_addr (bytes)
  g0[2] = (unsigned)ga;                                    // global_addr[31:0]
  g0[3] = ((unsigned)(ga >> 32) & 0x01FFFFFFu) | 0x80000000u; // [56:32] | type=2
  i32x8 g1;
  g1[0] = 0x00020000;          // workgroup_mask=0, data_size=2 (4 bytes)
  g1[1] = (int)(128u << 16);   // tensor_dim0 = 128 (lo16 of bits 79:48)
  g1[2] = (int)(8192u << 16);  // tensor_dim0 hi=0 | tensor_dim1 = 8192 (lo16)
  g1[3] = (int)(128u << 16);   // tensor_dim1 hi=0 | tile_dim0 = 128
  g1[4] = BC;                  // tile_dim1 = 64 | tile_dim2 = 0
  g1[5] = 128;                 // tensor_dim0_stride = 128 elements
  g1[6] = 0;                   // stride hi | tensor_dim1_stride lo
  g1[7] = 0;
  i32x4 z4 = {0, 0, 0, 0};
  i32x8 z8 = {0, 0, 0, 0, 0, 0, 0, 0};
  __builtin_amdgcn_tensor_load_to_lds(g0, g1, z4, z4, z8, 0);
}
#endif

__global__ __launch_bounds__(BLOCK) void fa_fwd_gfx1250(
    const float* __restrict__ Qg, const float* __restrict__ Kg,
    const float* __restrict__ Vg, float* __restrict__ Og)
{
  __shared__ unsigned short sK[BC][KPAD];      // K tile, row-major bf16
  __shared__ unsigned short sVT[DH][VPAD];     // V tile, transposed bf16: sVT[d][k]
  __shared__ unsigned short sP[WAVES][16][BC]; // per-wave P transpose scratch
#if HAS_TDM
  __shared__ __align__(16) float stageK[BC][DH];  // TDM fp32 staging
  __shared__ __align__(16) float stageV[BC][DH];
#endif

  const int tid  = threadIdx.x;
  const int wv   = tid >> 5;
  const int lane = tid & 31;
  const int hf   = lane >> 4;    // half-wave
  const int l16  = lane & 15;

#if HAS_TDM
  const unsigned offK = (unsigned)(size_t)&stageK[0][0];
  const unsigned offV = (unsigned)(size_t)&stageV[0][0];
  if (wv == 0) {                 // wave-level DMA: prefetch tile 0
    tdm_load_tile_f32(Kg, offK);
    tdm_load_tile_f32(Vg, offV);
  }
#endif

  // ---------- Q -> bf16 A-fragments (16x32 each, 4 chunks cover D=128) ----------
  // A layout: lane holds row m=l16; element e -> k = (e&7) + hf*8 + (e>>3)*16.
  Frag16 aq[4];
  {
    const int qrow = blockIdx.x * BR + wv * BQ + l16;
#pragma unroll
    for (int c = 0; c < 4; ++c) {
      const float* p = Qg + qrow * DH + c * 32 + hf * 8;
      f32x4 a0 = *(const f32x4*)(p + 0);
      f32x4 a1 = *(const f32x4*)(p + 4);
      f32x4 b0 = *(const f32x4*)(p + 16);
      f32x4 b1 = *(const f32x4*)(p + 20);
      u32x4 lo, hi;
      lo[0] = pack2(a0[0], a0[1]); lo[1] = pack2(a0[2], a0[3]);
      lo[2] = pack2(a1[0], a1[1]); lo[3] = pack2(a1[2], a1[3]);
      hi[0] = pack2(b0[0], b0[1]); hi[1] = pack2(b0[2], b0[3]);
      hi[2] = pack2(b1[0], b1[1]); hi[3] = pack2(b1[2], b1[3]);
      aq[c].q[0] = lo; aq[c].q[1] = hi;
    }
  }

  // ---------- accumulators / softmax state ----------
  v8f acc[8];                       // 8 x (16x16 f32 C-fragments) = 16 rows x 128 d
#pragma unroll
  for (int j = 0; j < 8; ++j)
#pragma unroll
    for (int r = 0; r < 8; ++r) acc[j][r] = 0.0f;
  float mrow[8], lrow[8];           // row m = hf*8 + r  (C-fragment row striping)
#pragma unroll
  for (int r = 0; r < 8; ++r) { mrow[r] = -1.0e30f; lrow[r] = 0.0f; }

  // cooperative conversion mappings
  const int kr  = tid >> 1;          // K: key row 0..63
  const int kc  = (tid & 1) * 64;    // K: 64-wide d segment
  const int vp2 = tid >> 2;          // V: key pair 0..31 (keys 2p,2p+1)
  const int vc  = (tid & 3) * 32;    // V: 32-wide d segment

#if HAS_TDM
  if (wv == 0) __builtin_amdgcn_s_wait_tensorcnt(0);  // tile 0 landed
  __syncthreads();                                    // broadcast arrival
#endif

  for (int kb = 0; kb < MTOT; kb += BC) {
#if HAS_TDM
    const float* srcK = &stageK[0][0];
    const float* srcV = &stageV[0][0];
#else
    __syncthreads();
    const float* srcK = Kg + kb * DH;
    const float* srcV = Vg + kb * DH;
#endif
    // ---------- convert fp32 tile -> bf16 sK (row-major) / sVT (transposed) -----
    {
      const float* kp = srcK + kr * DH + kc;
#pragma unroll
      for (int i = 0; i < 64; i += 8) {
        f32x4 x = *(const f32x4*)(kp + i);
        f32x4 y = *(const f32x4*)(kp + i + 4);
        u32x4 w;
        w[0] = pack2(x[0], x[1]); w[1] = pack2(x[2], x[3]);
        w[2] = pack2(y[0], y[1]); w[3] = pack2(y[2], y[3]);
        *(u32x4*)&sK[kr][kc + i] = w;
      }
      const float* va = srcV + (2 * vp2 + 0) * DH + vc;
      const float* vb = srcV + (2 * vp2 + 1) * DH + vc;
#pragma unroll
      for (int i = 0; i < 32; i += 4) {
        f32x4 x = *(const f32x4*)(va + i);
        f32x4 y = *(const f32x4*)(vb + i);
#pragma unroll
        for (int e = 0; e < 4; ++e)
          *(unsigned int*)&sVT[vc + i + e][2 * vp2] = pack2(x[e], y[e]);
      }
#if !HAS_TDM
      if (kb + BC < MTOT) {
        __builtin_prefetch(Kg + (kb + BC + kr) * DH + kc, 0, 0);
        __builtin_prefetch(Vg + (kb + BC + 2 * vp2) * DH + vc, 0, 0);
      }
#endif
    }
    __syncthreads();     // sK/sVT ready; staging buffer free for next DMA
#if HAS_TDM
    if (kb + BC < MTOT && wv == 0) {   // DMA tile t+1 overlaps compute of tile t
      tdm_load_tile_f32(Kg + (kb + BC) * DH, offK);
      tdm_load_tile_f32(Vg + (kb + BC) * DH, offV);
    }
#endif

    // ---------- S = Q K^T : four 16x16 tiles, K-depth 128 via 4 chained WMMAs ---
    // B layout: lane holds column n (key = ntile*16+l16); element e -> k = (e&7) + hf*8 + (e>>3)*16.
    v8f s[4];
#pragma unroll
    for (int n = 0; n < 4; ++n)
#pragma unroll
      for (int r = 0; r < 8; ++r) s[n][r] = 0.0f;
#pragma unroll
    for (int c = 0; c < 4; ++c) {
#pragma unroll
      for (int n = 0; n < 4; ++n) {
        Frag16 b;
        b.q[0] = *(const u32x4*)&sK[n * 16 + l16][c * 32 + hf * 8];
        b.q[1] = *(const u32x4*)&sK[n * 16 + l16][c * 32 + 16 + hf * 8];
        s[n] = __builtin_amdgcn_wmma_f32_16x16x32_bf16(false, aq[c].v, false, b.v,
                                                       (short)0, s[n], false, false);
      }
    }

    // ---------- online softmax over this 64-key block (VALU permlane reduce) ----
    float scale[8];
#pragma unroll
    for (int r = 0; r < 8; ++r) {
      float t = fmaxf(fmaxf(s[0][r], s[1][r]), fmaxf(s[2][r], s[3][r]));
      t = rowmax16(t);
      float mnew = fmaxf(mrow[r], t);
      float sc = __expf(mrow[r] - mnew);
      float p0 = __expf(s[0][r] - mnew);
      float p1 = __expf(s[1][r] - mnew);
      float p2 = __expf(s[2][r] - mnew);
      float p3 = __expf(s[3][r] - mnew);
      float ps = rowsum16((p0 + p1) + (p2 + p3));
      lrow[r] = lrow[r] * sc + ps;
      mrow[r] = mnew;
      scale[r] = sc;
      // stash P (C-layout: lane=key col, row m = hf*8+r) to per-wave LDS scratch
      sP[wv][hf * 8 + r][l16]      = bfu(p0);
      sP[wv][hf * 8 + r][16 + l16] = bfu(p1);
      sP[wv][hf * 8 + r][32 + l16] = bfu(p2);
      sP[wv][hf * 8 + r][48 + l16] = bfu(p3);
    }
#pragma unroll
    for (int j = 0; j < 8; ++j)
#pragma unroll
      for (int r = 0; r < 8; ++r) acc[j][r] *= scale[r];

    // LDS is in-order per wave; explicit dscnt drain before cross-lane re-read.
    asm volatile("s_wait_dscnt 0" ::: "memory");

    // ---------- O += P V : P in A-layout, 2 k-chunks x 8 D-tiles ----------
#pragma unroll
    for (int u = 0; u < 2; ++u) {
      Frag16 ap;   // element e -> key = u*32 + (e&7) + hf*8 + (e>>3)*16
      ap.q[0] = *(const u32x4*)&sP[wv][l16][u * 32 + hf * 8];
      ap.q[1] = *(const u32x4*)&sP[wv][l16][u * 32 + 16 + hf * 8];
#pragma unroll
      for (int j = 0; j < 8; ++j) {
        Frag16 bv;  // column n=l16 -> d=j*16+l16; rows k = u*32 + (e&7) + hf*8 + (e>>3)*16
        bv.q[0] = *(const u32x4*)&sVT[j * 16 + l16][u * 32 + hf * 8];
        bv.q[1] = *(const u32x4*)&sVT[j * 16 + l16][u * 32 + 16 + hf * 8];
        acc[j] = __builtin_amdgcn_wmma_f32_16x16x32_bf16(false, ap.v, false, bv.v,
                                                         (short)0, acc[j], false, false);
      }
    }

#if HAS_TDM
    if (wv == 0) __builtin_amdgcn_s_wait_tensorcnt(0);  // next tile landed
    __syncthreads();                                    // broadcast + sK reusable
#endif
  }

  // ---------- epilogue: divide by softmax denominator, store fp32 ----------
#pragma unroll
  for (int r = 0; r < 8; ++r) {
    float inv = 1.0f / lrow[r];
    float* op = Og + (blockIdx.x * BR + wv * BQ + hf * 8 + r) * DH + l16;
#pragma unroll
    for (int j = 0; j < 8; ++j) op[j * 16] = acc[j][r] * inv;
  }
}

extern "C" void kernel_launch(void* const* d_in, const int* in_sizes, int n_in,
                              void* d_out, int out_size, void* d_ws, size_t ws_size,
                              hipStream_t stream) {
  (void)in_sizes; (void)n_in; (void)d_ws; (void)ws_size; (void)out_size;
  const float* Q = (const float*)d_in[0];
  const float* K = (const float*)d_in[1];
  const float* V = (const float*)d_in[2];
  float* O = (float*)d_out;
  dim3 grid(MTOT / BR);   // 128 workgroups
  dim3 block(BLOCK);      // 128 threads = 4 waves
  hipLaunchKernelGGL(fa_fwd_gfx1250, grid, block, 0, stream, Q, K, V, O);
}
